// FlashMHA_14723147891227
// MI455X (gfx1250) — matrix operand order repaired and checked by reading, not compiled, not run on hardware
//
#include <hip/hip_runtime.h>
#include <hip/hip_bf16.h>

typedef __attribute__((ext_vector_type(16))) _Float16 v16h;
typedef __attribute__((ext_vector_type(8)))  _Float16 v8h;
typedef __attribute__((ext_vector_type(8)))  float    v8f;

#define EMBED   1024
#define NH      16
#define HD      64
#define BATCH   4
#define SEQ     2048
#define M_ROWS  (BATCH * SEQ)   /* 8192 */
#define F_QKV   (3 * EMBED)     /* 3072 */

// ---------------------------------------------------------------------------
// WMMA helpers (CDNA5 wave32, V_WMMA_F32_16X16X32_F16)
// ---------------------------------------------------------------------------
__device__ __forceinline__ v8f wmma_f16(v16h a, v16h b, v8f c) {
    return __builtin_amdgcn_wmma_f32_16x16x32_f16(
        /*neg_a=*/false, a, /*neg_b=*/false, b,
        /*c_mod=*/(short)0, c, /*reuse_a=*/false, /*reuse_b=*/false);
}

// A-frag: 16x32 f16 tile from row-major src (ld = row stride in elements).
// Lane L: row = L&15, K = (L>>4)*8 + {0..7} and (L>>4)*8 + 16 + {0..7}.
__device__ __forceinline__ v16h load_afrag(const _Float16* __restrict__ base,
                                           int ld, int row0, int k0, int lane) {
    const int m  = row0 + (lane & 15);
    const int kb = k0 + ((lane >> 4) << 3);
    const _Float16* p = base + (size_t)m * ld + kb;
    v16h r;
    ((v8h*)&r)[0] = *(const v8h*)(p);
    ((v8h*)&r)[1] = *(const v8h*)(p + 16);
    return r;
}

// B-frag: 32x16 f16 tile where B's columns are contiguous in memory
// (i.e. source matrix is (Ncols, K) row-major). Lane L: col = L&15,
// K = (L>>4)*8 + {0..7} and (L>>4)*8 + 16 + {0..7}.
__device__ __forceinline__ v16h load_bfrag(const _Float16* __restrict__ base,
                                           int ld, int n0, int k0, int lane) {
    const int n  = n0 + (lane & 15);
    const int kb = k0 + ((lane >> 4) << 3);
    const _Float16* p = base + (size_t)n * ld + kb;
    v16h r;
    ((v8h*)&r)[0] = *(const v8h*)(p);
    ((v8h*)&r)[1] = *(const v8h*)(p + 16);
    return r;
}

// ---------------------------------------------------------------------------
// fp32 -> fp16 conversion (one-time, ~75 MB total => ~3.2 us at 23.3 TB/s)
// ---------------------------------------------------------------------------
__global__ void f32_to_f16_kernel(const float* __restrict__ src,
                                  _Float16* __restrict__ dst, int n) {
    int i = blockIdx.x * blockDim.x + threadIdx.x;
    const int stride = gridDim.x * blockDim.x;
    for (; i < n; i += stride) dst[i] = (_Float16)src[i];
}

// ---------------------------------------------------------------------------
// QKV projection: qkv = x(8192x1024) @ w_qkv^T(1024x3072) + b_qkv
// Each wave owns a 64x64 output tile (16 accumulators). Epilogue scatters
// into Q,K (B,H,N,D) and V TRANSPOSED (B,H,D,N) so attention B-frags are
// contiguous.
// ---------------------------------------------------------------------------
__global__ __launch_bounds__(128) void qkv_gemm_kernel(
        const _Float16* __restrict__ X, const _Float16* __restrict__ W,
        const float* __restrict__ bias,
        _Float16* __restrict__ Qb, _Float16* __restrict__ Kb,
        _Float16* __restrict__ Vtb) {
    const int lane = threadIdx.x & 31;
    const int wave = threadIdx.x >> 5;                  // 0..3
    const int row0 = blockIdx.y * 128 + (wave >> 1) * 64;
    const int col0 = blockIdx.x * 128 + (wave & 1) * 64;

    v8f zero = {};
    v8f acc[4][4];
#pragma unroll
    for (int i = 0; i < 4; ++i)
#pragma unroll
        for (int j = 0; j < 4; ++j) acc[i][j] = zero;

    for (int k0 = 0; k0 < EMBED; k0 += 32) {
        v16h a[4], b[4];
#pragma unroll
        for (int i = 0; i < 4; ++i) a[i] = load_afrag(X, EMBED, row0 + 16 * i, k0, lane);
#pragma unroll
        for (int j = 0; j < 4; ++j) b[j] = load_bfrag(W, EMBED, col0 + 16 * j, k0, lane);
#pragma unroll
        for (int i = 0; i < 4; ++i)
#pragma unroll
            for (int j = 0; j < 4; ++j) acc[i][j] = wmma_f16(a[i], b[j], acc[i][j]);
    }

    const int g = lane >> 4, nl = lane & 15;
#pragma unroll
    for (int i = 0; i < 4; ++i) {
#pragma unroll
        for (int j = 0; j < 4; ++j) {
#pragma unroll
            for (int v = 0; v < 8; ++v) {
                const int m = row0 + 16 * i + v + 8 * g;   // row in [0,8192)
                const int f = col0 + 16 * j + nl;          // feature in [0,3072)
                const float val = acc[i][j][v] + bias[f];
                const _Float16 hv = (_Float16)val;
                const int which = f >> 10;                  // 0=Q 1=K 2=V
                const int c = f & 1023;
                const int h = c >> 6, d = c & 63;
                const int bb = m >> 11, n = m & 2047;
                const size_t bh = (size_t)(bb * NH + h);
                if (which == 0)      Qb [(bh * SEQ + n) * HD + d] = hv;
                else if (which == 1) Kb [(bh * SEQ + n) * HD + d] = hv;
                else                 Vtb[(bh * HD + d) * SEQ + n] = hv;  // transposed
            }
        }
    }
}

// ---------------------------------------------------------------------------
// Flash attention. Block = 8 waves; each wave owns 16 query rows and streams
// all 2048 keys in blocks of 32 with online softmax (exp2 domain).
// Per 32-key step: 4 score WMMAs + 4 PV WMMAs.
// ---------------------------------------------------------------------------
__global__ __launch_bounds__(256) void flash_attn_kernel(
        const _Float16* __restrict__ Qb, const _Float16* __restrict__ Kb,
        const _Float16* __restrict__ Vtb, _Float16* __restrict__ Oh) {
    __shared__ _Float16 Plds[8 * 16 * 32];              // 1 KB per wave

    const int lane = threadIdx.x & 31;
    const int wave = threadIdx.x >> 5;                  // 0..7
    const int bh   = blockIdx.y;                        // 0..63
    const int b    = bh >> 4, h = bh & 15;
    const int qbase = blockIdx.x * 128 + wave * 16;
    const int g = lane >> 4, nl = lane & 15;

    const _Float16* Qp = Qb  + (size_t)bh * SEQ * HD;
    const _Float16* Kp = Kb  + (size_t)bh * SEQ * HD;
    const _Float16* Vp = Vtb + (size_t)bh * HD * SEQ;   // (D, N) layout
    _Float16* Pw = Plds + wave * 512;

    v16h qf[2];
    qf[0] = load_afrag(Qp, HD, qbase, 0, lane);
    qf[1] = load_afrag(Qp, HD, qbase, 32, lane);

    v8f zero = {};
    v8f accO[4];
#pragma unroll
    for (int c = 0; c < 4; ++c) accO[c] = zero;
    float mrow[8], lrow[8];
#pragma unroll
    for (int v = 0; v < 8; ++v) { mrow[v] = -1e30f; lrow[v] = 0.0f; }

    const float SCL = 0.18033688011112042f;             // log2(e) / sqrt(64)

    for (int j0 = 0; j0 < SEQ; j0 += 32) {
        // scores S = Q @ K^T for 32 keys (two 16x16 tiles)
        v8f s0 = zero, s1 = zero;
#pragma unroll
        for (int k = 0; k < 2; ++k) {
            const v16h bk0 = load_bfrag(Kp, HD, j0,      32 * k, lane);
            const v16h bk1 = load_bfrag(Kp, HD, j0 + 16, 32 * k, lane);
            s0 = wmma_f16(qf[k], bk0, s0);
            s1 = wmma_f16(qf[k], bk1, s1);
        }

        float p0[8], p1[8];
#pragma unroll
        for (int v = 0; v < 8; ++v) {
            const float t0 = s0[v] * SCL, t1 = s1[v] * SCL;
            float r = fmaxf(t0, t1);                     // row-max over 16 lanes
            r = fmaxf(r, __shfl_xor(r, 1));
            r = fmaxf(r, __shfl_xor(r, 2));
            r = fmaxf(r, __shfl_xor(r, 4));
            r = fmaxf(r, __shfl_xor(r, 8));
            const float mn = fmaxf(mrow[v], r);
            const float alpha = exp2f(mrow[v] - mn);
            p0[v] = exp2f(t0 - mn);
            p1[v] = exp2f(t1 - mn);
            float rs = p0[v] + p1[v];
            rs += __shfl_xor(rs, 1);
            rs += __shfl_xor(rs, 2);
            rs += __shfl_xor(rs, 4);
            rs += __shfl_xor(rs, 8);
            lrow[v] = lrow[v] * alpha + rs;
            mrow[v] = mn;
#pragma unroll
            for (int c = 0; c < 4; ++c) accO[c][v] *= alpha;
        }

        // C-layout -> A-layout transpose of P through per-wave LDS
#pragma unroll
        for (int v = 0; v < 8; ++v) {
            const int m = v + 8 * g;
            Pw[m * 32 + nl]      = (_Float16)p0[v];
            Pw[m * 32 + 16 + nl] = (_Float16)p1[v];
        }
        v16h pf;                                         // same-wave DS ops are in order
        {
            const _Float16* pp = Pw + nl * 32 + g * 8;
            ((v8h*)&pf)[0] = *(const v8h*)(pp);
            ((v8h*)&pf)[1] = *(const v8h*)(pp + 16);
        }

        // O += P @ V  (V pre-transposed => contiguous B-frags)
#pragma unroll
        for (int c = 0; c < 4; ++c) {
            const v16h bv = load_bfrag(Vp, SEQ, c * 16, j0, lane);
            accO[c] = wmma_f16(pf, bv, accO[c]);
        }
    }

    // normalize + store (B, N, C) so out-proj reads plain row-major A
#pragma unroll
    for (int v = 0; v < 8; ++v) {
        const float inv = 1.0f / lrow[v];
        const int m = v + 8 * g;
        const int n = qbase + m;
#pragma unroll
        for (int c = 0; c < 4; ++c) {
            const int col = h * 64 + c * 16 + nl;
            Oh[((size_t)b * SEQ + n) * EMBED + col] = (_Float16)(accO[c][v] * inv);
        }
    }
}

// ---------------------------------------------------------------------------
// Output projection: out = O(8192x1024) @ w_o^T(1024x1024) + b_o, fp32 out
// ---------------------------------------------------------------------------
__global__ __launch_bounds__(128) void out_gemm_kernel(
        const _Float16* __restrict__ A, const _Float16* __restrict__ W,
        const float* __restrict__ bias, float* __restrict__ out) {
    const int lane = threadIdx.x & 31;
    const int wave = threadIdx.x >> 5;
    const int row0 = blockIdx.y * 128 + (wave >> 1) * 64;
    const int col0 = blockIdx.x * 128 + (wave & 1) * 64;

    v8f zero = {};
    v8f acc[4][4];
#pragma unroll
    for (int i = 0; i < 4; ++i)
#pragma unroll
        for (int j = 0; j < 4; ++j) acc[i][j] = zero;

    for (int k0 = 0; k0 < EMBED; k0 += 32) {
        v16h a[4], b[4];
#pragma unroll
        for (int i = 0; i < 4; ++i) a[i] = load_afrag(A, EMBED, row0 + 16 * i, k0, lane);
#pragma unroll
        for (int j = 0; j < 4; ++j) b[j] = load_bfrag(W, EMBED, col0 + 16 * j, k0, lane);
#pragma unroll
        for (int i = 0; i < 4; ++i)
#pragma unroll
            for (int j = 0; j < 4; ++j) acc[i][j] = wmma_f16(a[i], b[j], acc[i][j]);
    }

    const int g = lane >> 4, nl = lane & 15;
#pragma unroll
    for (int i = 0; i < 4; ++i)
#pragma unroll
        for (int j = 0; j < 4; ++j)
#pragma unroll
            for (int v = 0; v < 8; ++v) {
                const int m = row0 + 16 * i + v + 8 * g;
                const int f = col0 + 16 * j + nl;
                out[(size_t)m * EMBED + f] = acc[i][j][v] + bias[f];
            }
}

// ---------------------------------------------------------------------------
// Host launcher
// ---------------------------------------------------------------------------
extern "C" void kernel_launch(void* const* d_in, const int* in_sizes, int n_in,
                              void* d_out, int out_size, void* d_ws, size_t ws_size,
                              hipStream_t stream) {
    (void)in_sizes; (void)n_in; (void)out_size; (void)ws_size;
    const float* x     = (const float*)d_in[0];
    const float* w_qkv = (const float*)d_in[1];
    const float* b_qkv = (const float*)d_in[2];
    const float* w_o   = (const float*)d_in[3];
    const float* b_o   = (const float*)d_in[4];

    char* ws = (char*)d_ws;
    size_t off = 0;
    auto carve = [&](size_t bytes) -> void* {
        void* p = ws + off;
        off += (bytes + 255) & ~(size_t)255;
        return p;
    };
    _Float16* x_h    = (_Float16*)carve((size_t)M_ROWS * EMBED * 2);
    _Float16* wqkv_h = (_Float16*)carve((size_t)F_QKV * EMBED * 2);
    _Float16* wo_h   = (_Float16*)carve((size_t)EMBED * EMBED * 2);
    _Float16* Qb     = (_Float16*)carve((size_t)BATCH * NH * SEQ * HD * 2);
    _Float16* Kb     = (_Float16*)carve((size_t)BATCH * NH * SEQ * HD * 2);
    _Float16* Vtb    = (_Float16*)carve((size_t)BATCH * NH * SEQ * HD * 2);
    _Float16* Oh     = (_Float16*)carve((size_t)M_ROWS * EMBED * 2);
    // total workspace: ~88 MB

    f32_to_f16_kernel<<<4096, 256, 0, stream>>>(x,     x_h,    M_ROWS * EMBED);
    f32_to_f16_kernel<<<2048, 256, 0, stream>>>(w_qkv, wqkv_h, F_QKV * EMBED);
    f32_to_f16_kernel<<<1024, 256, 0, stream>>>(w_o,   wo_h,   EMBED * EMBED);

    // 8192x3072 output, 128x128 block tiles
    qkv_gemm_kernel<<<dim3(F_QKV / 128, M_ROWS / 128), 128, 0, stream>>>(
        x_h, wqkv_h, b_qkv, Qb, Kb, Vtb);

    // grid: (q-blocks of 128, B*H)
    flash_attn_kernel<<<dim3(SEQ / 128, BATCH * NH), 256, 0, stream>>>(Qb, Kb, Vtb, Oh);

    // 8192x1024 output
    out_gemm_kernel<<<dim3(EMBED / 128, M_ROWS / 128), 128, 0, stream>>>(
        Oh, wo_h, b_o, (float*)d_out);
}
